// TemporalFusionTransformer_68307159875978
// MI455X (gfx1250) — matrix operand order repaired and checked by reading, not compiled, not run on hardware
//
#include <hip/hip_runtime.h>
#include <hip/hip_bf16.h>

// ---------------------------------------------------------------------------
// TemporalFusionTransformer for MI455X (gfx1250).
// Dense GEMMs: LDS-staged, block-cooperative v_wmma_f32_16x16x32_f16
// (64x128 block tile, 64x16 per wave => 4 WMMA per K-step per wave).
// Sliding-window attention + layernorm as VALU kernels.
// ---------------------------------------------------------------------------

#define BDIM   4
#define SLEN   2048
#define DIN    64
#define DM     256
#define NHEAD  8
#define HD     32
#define WSZ    65
#define NL     3
#define NTOK   (BDIM * SLEN)          // 8192

#define ASTRIDE 40                    // padded LDS row stride (halves) for A

typedef __attribute__((ext_vector_type(16))) _Float16 v16h;
typedef __attribute__((ext_vector_type(8)))  _Float16 v8h;
typedef __attribute__((ext_vector_type(8)))  float    v8f;

// ---------------------------------------------------------------------------
// f32 -> f16 weight conversion
// ---------------------------------------------------------------------------
__global__ void cvt_f32_to_f16(const float* __restrict__ src,
                               _Float16* __restrict__ dst, int n) {
    int i = blockIdx.x * blockDim.x + threadIdx.x;
    if (i < n) dst[i] = (_Float16)src[i];
}

// ---------------------------------------------------------------------------
// WMMA GEMM: out[M,N] = act(A[M,K] @ W[K,N] + bias[N])
//   act 0: identity   act 1: relu   act 2: out = gate[m,n] * sigmoid(.)
// Block tile: 64 rows x 128 cols, 8 waves; wave w owns cols [16w, 16w+16)
// and all 4 row sub-tiles (4 accumulators, 4 WMMA per K-step).
// A slab (64x32 f32 -> f16) and B slab (32x128 f16) staged in LDS each step.
// Requires: M % 64 == 0, K % 32 == 0, N % 128 == 0 (true for every call).
// ---------------------------------------------------------------------------
__global__ __launch_bounds__(256)
void wmma_gemm_kernel(const float* __restrict__ A, const _Float16* __restrict__ W,
                      const float* __restrict__ bias, const float* __restrict__ gate,
                      float* __restrict__ out, int M, int K, int N, int act) {
    __shared__ _Float16 As[64 * ASTRIDE];   // 5 KB
    __shared__ _Float16 Bs[32 * 128];       // 8 KB

    const int t     = threadIdx.x;
    const int lane  = t & 31;
    const int wave  = t >> 5;
    const int lr    = lane & 15;            // row-in-tile (A) / col-in-tile (B,C)
    const int hk    = lane >> 4;            // half-wave selector
    const int tileM = blockIdx.y * 64;
    const int colB  = blockIdx.x * 128;     // block's first output column

    // cooperative-load index precompute
    const int arow = t >> 2;                // 0..63
    const int acol = (t & 3) * 8;           // 0,8,16,24
    const int brow = t >> 3;                // 0..31
    const int bcol = (t & 7) * 16;          // 0..112

    v8f acc[4] = {};

    for (int k0 = 0; k0 < K; k0 += 32) {
        // ---- stage A slab: 64x32 f32 -> f16, coalesced b128 loads ----
        {
            const float* ag = A + (size_t)(tileM + arow) * K + k0 + acol;
            float4 f0 = ((const float4*)ag)[0];
            float4 f1 = ((const float4*)ag)[1];
            v8h hv;
            hv[0] = (_Float16)f0.x; hv[1] = (_Float16)f0.y;
            hv[2] = (_Float16)f0.z; hv[3] = (_Float16)f0.w;
            hv[4] = (_Float16)f1.x; hv[5] = (_Float16)f1.y;
            hv[6] = (_Float16)f1.z; hv[7] = (_Float16)f1.w;
            *(v8h*)(As + arow * ASTRIDE + acol) = hv;
        }
        // ---- stage B slab: 32x128 f16, coalesced 32B loads ----
        {
            const _Float16* bg = W + (size_t)(k0 + brow) * N + colB + bcol;
            v8h b0 = ((const v8h*)bg)[0];
            v8h b1 = ((const v8h*)bg)[1];
            _Float16* bs = Bs + brow * 128 + bcol;
            ((v8h*)bs)[0] = b0;
            ((v8h*)bs)[1] = b1;
        }
        __syncthreads();

        // ---- B fragment (32x16): col = 16*wave + lr, K = 8*hk + e + (e & 8) ----
        v16h b;
        {
            const _Float16* bp = Bs + (8 * hk) * 128 + wave * 16 + lr;
#pragma unroll
            for (int e = 0; e < 16; ++e) b[e] = bp[(e + (e & 8)) * 128];
        }
        // ---- 4 A fragments + 4 WMMA ----
#pragma unroll
        for (int m = 0; m < 4; ++m) {
            const _Float16* ap = As + (16 * m + lr) * ASTRIDE + 8 * hk;
            v8h a0 = *(const v8h*)ap;
            v8h a1 = *(const v8h*)(ap + 16);
            v16h a;
#pragma unroll
            for (int e = 0; e < 8; ++e) { a[e] = a0[e]; a[8 + e] = a1[e]; }
            acc[m] = __builtin_amdgcn_wmma_f32_16x16x32_f16(
                        false, a, false, b, (short)0, acc[m], false, false);
        }
        __syncthreads();
    }

    // ---- epilogue: vgpr j of subtile m -> row tileM + 16m + j + 8*hk ----
    const int col  = colB + wave * 16 + lr;
    const float bc = bias[col];
#pragma unroll
    for (int m = 0; m < 4; ++m) {
#pragma unroll
        for (int j = 0; j < 8; ++j) {
            const int row = tileM + 16 * m + j + 8 * hk;
            float v = acc[m][j] + bc;
            if (act == 1) {
                v = fmaxf(v, 0.f);
            } else if (act == 2) {
                v = gate[(size_t)row * N + col] *
                    __builtin_amdgcn_rcpf(1.f + __expf(-v));
            }
            out[(size_t)row * N + col] = v;
        }
    }
}

// ---------------------------------------------------------------------------
// Sliding-window attention (zero-padded like the reference: OOB score == 0,
// still inside the softmax). One thread per (b, h, s).
//   qkv: [NTOK, 768] token-major ([3, H, 32] inside)
//   aw : this layer's slice of d_out, layout (B, H, S, WSZ)
//   att: [NTOK, 256] per-head outputs, pre-projection
// ---------------------------------------------------------------------------
__global__ __launch_bounds__(256)
void swa_kernel(const float* __restrict__ qkv, float* __restrict__ aw,
                float* __restrict__ att) {
    const int idx = blockIdx.x * blockDim.x + threadIdx.x;  // (b*H + h)*S + s
    if (idx >= BDIM * NHEAD * SLEN) return;
    const int s  = idx & (SLEN - 1);
    const int bh = idx >> 11;          // b*H + h
    const int h  = bh & (NHEAD - 1);
    const int b  = bh >> 3;

    const float* qr = qkv + (size_t)(b * SLEN + s) * (3 * DM) + h * HD;
    float q[HD];
#pragma unroll
    for (int d = 0; d < HD; ++d) q[d] = qr[d];

    float sc[WSZ];
    float mx = -1e30f;
    for (int w = 0; w < WSZ; ++w) {
        const int j = s + w - (WSZ / 2);
        float t = 0.f;
        if (j >= 0 && j < SLEN) {
            const float* kr = qkv + (size_t)(b * SLEN + j) * (3 * DM) + DM + h * HD;
#pragma unroll
            for (int d = 0; d < HD; ++d) t += q[d] * kr[d];
            t *= 0.17677669529663687f;  // 1/sqrt(32)
        }
        sc[w] = t;
        mx = fmaxf(mx, t);
    }
    float sum = 0.f;
    for (int w = 0; w < WSZ; ++w) {
        const float e = __expf(sc[w] - mx);
        sc[w] = e;
        sum += e;
    }
    const float inv = __builtin_amdgcn_rcpf(sum);

    float o[HD];
#pragma unroll
    for (int d = 0; d < HD; ++d) o[d] = 0.f;

    float* awr = aw + (size_t)idx * WSZ;  // (B,H,S,W) flat == idx*WSZ
    for (int w = 0; w < WSZ; ++w) {
        const float a = sc[w] * inv;
        awr[w] = a;
        const int j = s + w - (WSZ / 2);
        if (j >= 0 && j < SLEN) {
            const float* vr = qkv + (size_t)(b * SLEN + j) * (3 * DM) + 2 * DM + h * HD;
#pragma unroll
            for (int d = 0; d < HD; ++d) o[d] += a * vr[d];
        }
    }
    float* orow = att + (size_t)(b * SLEN + s) * DM + h * HD;
#pragma unroll
    for (int d = 0; d < HD; ++d) orow[d] = o[d];
}

// ---------------------------------------------------------------------------
// h = LayerNorm(h + add) * g + be   (in place on h). One wave32 per token,
// 8 elements/lane, shuffle reductions.
// ---------------------------------------------------------------------------
__global__ __launch_bounds__(256)
void ln_residual_kernel(float* __restrict__ h, const float* __restrict__ add,
                        const float* __restrict__ g, const float* __restrict__ be) {
    const int lane  = threadIdx.x & 31;
    const int wave  = threadIdx.x >> 5;
    const int token = blockIdx.x * 8 + wave;
    if (token >= NTOK) return;

    float* hr       = h   + (size_t)token * DM;
    const float* ar = add + (size_t)token * DM;

    float v[8];
    float s = 0.f;
#pragma unroll
    for (int i = 0; i < 8; ++i) {
        v[i] = hr[lane * 8 + i] + ar[lane * 8 + i];
        s += v[i];
    }
#pragma unroll
    for (int off = 16; off > 0; off >>= 1) s += __shfl_xor(s, off, 32);
    const float mean = s * (1.f / DM);

    float vs = 0.f;
#pragma unroll
    for (int i = 0; i < 8; ++i) {
        const float d = v[i] - mean;
        vs += d * d;
    }
#pragma unroll
    for (int off = 16; off > 0; off >>= 1) vs += __shfl_xor(vs, off, 32);
    const float rstd = rsqrtf(vs * (1.f / DM) + 1e-5f);

#pragma unroll
    for (int i = 0; i < 8; ++i) {
        const int c = lane * 8 + i;
        hr[c] = (v[i] - mean) * rstd * g[c] + be[c];
    }
}

// ---------------------------------------------------------------------------
// Final head: scores[t] = relu_act[t, 0:128] . Wh2 + bh2   (N==1 GEMM)
// ---------------------------------------------------------------------------
__global__ __launch_bounds__(256)
void head2_kernel(const float* __restrict__ act, const float* __restrict__ Wh2,
                  const float* __restrict__ bh2, float* __restrict__ out) {
    const int t = blockIdx.x * blockDim.x + threadIdx.x;
    if (t >= NTOK) return;
    const float* r = act + (size_t)t * 128;
    float s = bh2[0];
#pragma unroll 4
    for (int i = 0; i < 128; ++i) s += r[i] * Wh2[i];
    out[t] = s;
}

// ---------------------------------------------------------------------------
// Host launcher
// ---------------------------------------------------------------------------
extern "C" void kernel_launch(void* const* d_in, const int* in_sizes, int n_in,
                              void* d_out, int out_size, void* d_ws, size_t ws_size,
                              hipStream_t stream) {
    (void)in_sizes; (void)n_in; (void)out_size; (void)ws_size;

    const float* x   = (const float*)d_in[0];
    const float* Win = (const float*)d_in[1];
    const float* bin = (const float*)d_in[2];
    const float* Wg  = (const float*)d_in[3];
    const float* bg  = (const float*)d_in[4];
    // layer L: base 5 + 12*L : Wqkv,bqkv,Wp,bp,W1,b1,W2,b2,g1,be1,g2,be2
    const float* Wh1 = (const float*)d_in[41];
    const float* bh1 = (const float*)d_in[42];
    const float* Wh2 = (const float*)d_in[43];
    const float* bh2 = (const float*)d_in[44];

    float* out = (float*)d_out;
    float* scores_out = out;                       // [NTOK] floats
    float* aws_out    = out + NTOK;                // [NL,B,H,S,WSZ]
    const size_t AW_L = (size_t)BDIM * NHEAD * SLEN * WSZ;

    // ---- workspace layout (floats) ----
    float* ws = (float*)d_ws;
    const size_t HSZ = (size_t)NTOK * DM;          // 2,097,152
    float* hA  = ws;                               // scratch / gate source
    float* hB  = hA + HSZ;                         // persistent hidden state
    float* big = hB + HSZ;                         // [NTOK, 1024] qkv / ffn-hidden
    float* att = big + (size_t)NTOK * 1024;        // [NTOK, 256]
    _Float16* w16 = (_Float16*)(att + HSZ);        // f16 weight pool

    _Float16* Win16 = w16;                                   // 64*256
    _Float16* Wg16  = Win16 + (size_t)DIN * DM;              // 256*256
    _Float16* lay16 = Wg16 + (size_t)DM * DM;
    const size_t LSTRIDE = (size_t)DM * (3 * DM)             // Wqkv
                         + (size_t)DM * DM                   // Wp
                         + (size_t)DM * (4 * DM)             // W1
                         + (size_t)(4 * DM) * DM;            // W2
    _Float16* Wh116 = lay16 + NL * LSTRIDE;                  // 256*128

    auto cvt = [&](const float* src, _Float16* dst, int n) {
        cvt_f32_to_f16<<<(n + 255) / 256, 256, 0, stream>>>(src, dst, n);
    };
    auto gemm = [&](const float* A, const _Float16* W, const float* bias,
                    const float* gate, float* o, int M, int K, int N, int act) {
        dim3 grd(N / 128, M / 64);
        wmma_gemm_kernel<<<grd, 256, 0, stream>>>(A, W, bias, gate, o, M, K, N, act);
    };

    // ---- convert all weights to f16 once ----
    cvt(Win, Win16, DIN * DM);
    cvt(Wg,  Wg16,  DM * DM);
    for (int L = 0; L < NL; ++L) {
        const float* Wqkv = (const float*)d_in[5 + 12 * L + 0];
        const float* Wp   = (const float*)d_in[5 + 12 * L + 2];
        const float* W1   = (const float*)d_in[5 + 12 * L + 4];
        const float* W2   = (const float*)d_in[5 + 12 * L + 6];
        _Float16* base = lay16 + L * LSTRIDE;
        cvt(Wqkv, base,                                         DM * 3 * DM);
        cvt(Wp,   base + (size_t)DM * 3 * DM,                   DM * DM);
        cvt(W1,   base + (size_t)DM * 3 * DM + (size_t)DM * DM, DM * 4 * DM);
        cvt(W2,   base + (size_t)DM * 3 * DM + (size_t)DM * DM + (size_t)DM * 4 * DM,
                  4 * DM * DM);
    }
    cvt(Wh1, Wh116, DM * 128);

    // ---- input projection + sigmoid gate ----
    gemm(x,  Win16, bin, nullptr, hA, NTOK, DIN, DM, 0);   // hA = x@Win + bin
    gemm(hA, Wg16,  bg,  hA,      hB, NTOK, DM,  DM, 2);   // hB = hA * sigmoid(hA@Wg+bg)

    // ---- transformer layers (state lives in hB) ----
    for (int L = 0; L < NL; ++L) {
        const float* bqkv = (const float*)d_in[5 + 12 * L + 1];
        const float* bp   = (const float*)d_in[5 + 12 * L + 3];
        const float* b1   = (const float*)d_in[5 + 12 * L + 5];
        const float* b2   = (const float*)d_in[5 + 12 * L + 7];
        const float* g1   = (const float*)d_in[5 + 12 * L + 8];
        const float* be1  = (const float*)d_in[5 + 12 * L + 9];
        const float* g2   = (const float*)d_in[5 + 12 * L + 10];
        const float* be2  = (const float*)d_in[5 + 12 * L + 11];
        _Float16* base   = lay16 + L * LSTRIDE;
        _Float16* Wqkv16 = base;
        _Float16* Wp16   = base + (size_t)DM * 3 * DM;
        _Float16* W116   = Wp16 + (size_t)DM * DM;
        _Float16* W216   = W116 + (size_t)DM * 4 * DM;

        // qkv = h @ Wqkv + bqkv
        gemm(hB, Wqkv16, bqkv, nullptr, big, NTOK, DM, 3 * DM, 0);
        // sliding-window attention -> att heads, aw into d_out slice
        swa_kernel<<<(BDIM * NHEAD * SLEN) / 256, 256, 0, stream>>>(
            big, aws_out + L * AW_L, att);
        // projection: hA = att @ Wp + bp
        gemm(att, Wp16, bp, nullptr, hA, NTOK, DM, DM, 0);
        // h = LN(h + attn_proj)
        ln_residual_kernel<<<NTOK / 8, 256, 0, stream>>>(hB, hA, g1, be1);
        // FFN
        gemm(hB,  W116, b1, nullptr, big, NTOK, DM,     4 * DM, 1);  // relu
        gemm(big, W216, b2, nullptr, hA,  NTOK, 4 * DM, DM,     0);
        // h = LN(h + ffn)
        ln_residual_kernel<<<NTOK / 8, 256, 0, stream>>>(hB, hA, g2, be2);
    }

    // ---- head ----
    gemm(hB, Wh116, bh1, nullptr, att, NTOK, DM, 128, 1);            // relu, [NTOK,128]
    head2_kernel<<<NTOK / 256, 256, 0, stream>>>(att, Wh2, bh2, scores_out);
}
